// CUTSEncoder_37297495998961
// MI455X (gfx1250) — matrix operand order repaired and checked by reading, not compiled, not run on hardware
//
#include <hip/hip_runtime.h>

typedef __attribute__((ext_vector_type(16))) _Float16 v16h;
typedef __attribute__((ext_vector_type(8)))  _Float16 v8h;
typedef __attribute__((ext_vector_type(8)))  float    v8f;

static constexpr int   BN_  = 4;
static constexpr int   CIN0 = 3;
static constexpr int   H_   = 384;
static constexpr int   W_   = 384;
static constexpr int   HW_  = H_ * W_;
static constexpr int   PP_  = 9;      // patch size
static constexpr int   HALFP = 4;
static constexpr int   NS_  = 4;
static constexpr int   WTILE = 128;   // pixels per block strip
static constexpr int   NBLK  = BN_ * H_ * (W_ / WTILE);   // 4608 conv blocks
static constexpr float EPS_   = 1e-5f;
static constexpr float SLOPE_ = 0.01f;

// ---------------------------------------------------------------------------
// Weight pre-pack: OIHW fp32 -> f16 in WMMA-B fragment order, with K
// enumerated as k' = (r*3+s)*CIN + c  (channel-innermost) so the conv kernel
// can fetch A fragments as contiguous 16-byte LDS vectors.
// B (32x16 f16): lane l holds column n=l%16; v16h element i holds
// k' = (l/16)*8 + (i < 8 ? i : i + 8) within each 32-wide K block.
// Flat layout: packed[((ct*KSTEPS+ks)*32 + lane)*16 + i]
// ---------------------------------------------------------------------------
__global__ void pack_weights(const float* __restrict__ w, _Float16* __restrict__ packed,
                             int COUT, int CIN, int KSTEPS)
{
    int K = CIN * 9;
    int total = (COUT / 16) * KSTEPS * 512;
    int idx = blockIdx.x * blockDim.x + threadIdx.x;
    if (idx >= total) return;
    int i    = idx & 15;
    int lane = (idx >> 4) & 31;
    int ks   = (idx >> 9) % KSTEPS;
    int ct   = (idx >> 9) / KSTEPS;
    int kp = ks * 32 + (lane >> 4) * 8 + (i < 8 ? i : i + 8);     // k' (channel-innermost order)
    int o  = ct * 16 + (lane & 15);
    float v = 0.0f;
    if (kp < K) {
        int rs = kp / CIN;                       // r*3+s
        int c  = kp - rs * CIN;
        v = w[(size_t)o * K + c * 9 + rs];       // OIHW flat = o*CIN*9 + c*9 + rs
    }
    packed[idx] = (_Float16)v;
}

__global__ void cast_f32_to_f16(const float* __restrict__ src,
                                _Float16* __restrict__ dst, size_t n)
{
    size_t i = (size_t)blockIdx.x * blockDim.x + threadIdx.x;
    if (i < n) dst[i] = (_Float16)src[i];
}

// ---------------------------------------------------------------------------
// Implicit-GEMM 3x3 SAME conv + bias, f16 in / f16 out, f32 WMMA accumulate.
// LDS strip layout: s_in[r][col][CINP] (channel-innermost, CINP = CIN+8 pad)
// so one A-fragment half (8 consecutive k') = one ds_load_b128.
// Block: 256 threads (8 waves). Strip: one (n,h) row, 128 pixels.
// Wave w owns pixel tile w (16 px), loops over COUT/16 column tiles.
// ---------------------------------------------------------------------------
template<int CIN, int COUT>
__global__ __launch_bounds__(256)
void conv_wmma(const _Float16* __restrict__ act, const _Float16* __restrict__ wpack,
               const float* __restrict__ bias, _Float16* __restrict__ y,
               float* __restrict__ psum, float* __restrict__ psq)
{
    constexpr int  K       = CIN * 9;
    constexpr int  KSTEPS  = (K + 31) / 32;
    constexpr int  CT      = COUT / 16;
    constexpr bool VECA    = (CIN % 8 == 0);
    constexpr int  CINP    = VECA ? CIN + 8 : CIN;        // pad: alignment + banks
    constexpr int  LOG2C   = (CIN == 16) ? 4 : (CIN == 32) ? 5 : 6;
    constexpr int  COLS    = WTILE + 2;
    constexpr size_t SM_IN   = (size_t)3 * COLS * CINP * sizeof(_Float16);
    constexpr size_t SM_PART = (size_t)COUT * 16 * 2 * sizeof(float);
    constexpr size_t SMEM    = SM_IN > SM_PART ? SM_IN : SM_PART;

    __shared__ __align__(16) char smem[SMEM];
    _Float16* s_inF = (_Float16*)smem;         // [(r*COLS + col)*CINP + c]
    float* s_psum = (float*)smem;              // [COUT][16] after A-frags captured
    float* s_psq  = s_psum + COUT * 16;

    const int tid = threadIdx.x;
    const int n = blockIdx.z, h = blockIdx.y, w0 = blockIdx.x * WTILE;

    // Stage input strip + halo (cols w0-1 .. w0+128, rows h-1..h+1) into LDS.
    // Enumerate (c, r, col) so global reads stay coalesced along col.
    for (int idx = tid; idx < CIN * 3 * COLS; idx += 256) {
        int col = idx % COLS;
        int r   = (idx / COLS) % 3;
        int c   = idx / (3 * COLS);
        int gh = h - 1 + r;
        int gw = w0 - 1 + col;
        _Float16 v = (_Float16)0.0f;
        if ((unsigned)gh < (unsigned)H_ && (unsigned)gw < (unsigned)W_)
            v = act[(((size_t)n * CIN + c) * H_ + gh) * W_ + gw];
        s_inF[(r * COLS + col) * CINP + c] = v;
    }
    __syncthreads();

    const int wave  = tid >> 5;
    const int lane  = tid & 31;
    const int m     = lane & 15;                 // A row / B,C,D column
    const int px    = wave * 16 + m;             // pixel within strip
    const int khalf = (lane < 16) ? 0 : 8;
    const int hi    = (lane < 16) ? 0 : 1;
    const int mrow  = hi * 8;                    // C/D row offset for this lane

    // A fragments (16x32 f16), shared across every column tile of this wave.
    v16h afrag[KSTEPS];
    #pragma unroll
    for (int ks = 0; ks < KSTEPS; ++ks) {
        if constexpr (VECA) {
            #pragma unroll
            for (int half = 0; half < 2; ++half) {
                const int base = ks * 32 + khalf + half * 16;  // multiple of 8
                v8h part = {};
                if ((K % 32 == 0) || (base < K)) {
                    int rs = base >> LOG2C;                    // r*3+s
                    int c0 = base & (CIN - 1);                 // first channel
                    int r  = rs / 3;
                    int s  = rs - 3 * r;
                    part = *(const v8h*)&s_inF[(r * COLS + (px + s)) * CINP + c0];
                }
                #pragma unroll
                for (int i = 0; i < 8; ++i) afrag[ks][half * 8 + i] = part[i];
            }
        } else {
            #pragma unroll
            for (int i = 0; i < 16; ++i) {
                int kp = ks * 32 + khalf + (i < 8 ? i : i + 8);
                _Float16 v = (_Float16)0.0f;
                if (kp < K) {
                    int rs = kp / CIN;
                    int c  = kp - rs * CIN;
                    int r  = rs / 3, s = rs - 3 * r;
                    v = s_inF[(r * COLS + (px + s)) * CINP + c];
                }
                afrag[ks][i] = v;
            }
        }
    }
    __syncthreads();          // s_in dead; smem now reused for BN partials

    #pragma unroll
    for (int ct = 0; ct < CT; ++ct) {
        v8f acc = {};
        #pragma unroll
        for (int ks = 0; ks < KSTEPS; ++ks) {
            const v16h b = *(const v16h*)(wpack +
                (((size_t)(ct * KSTEPS + ks)) * 32 + lane) * 16);
            acc = __builtin_amdgcn_wmma_f32_16x16x32_f16(
                false, afrag[ks], false, b, (short)0, acc, false, false);
        }
        const int o  = ct * 16 + m;              // output channel of this lane
        const float bo = bias[o];
        float lsum = 0.0f, lsq = 0.0f;
        v8h out8;
        #pragma unroll
        for (int r = 0; r < 8; ++r) {
            float v = acc[r] + bo;
            lsum += v; lsq += v * v;
            out8[r] = (_Float16)v;
        }
        s_psum[o * 16 + wave * 2 + hi] = lsum;   // unique slot: no atomics
        s_psq [o * 16 + wave * 2 + hi] = lsq;
        *(v8h*)(y + ((((size_t)n * COUT + o) * H_ + h) * W_ + w0 + wave * 16 + mrow)) = out8;
    }
    __syncthreads();

    if (tid < COUT) {                            // fixed-order, deterministic
        float a = 0.0f, b = 0.0f;
        #pragma unroll
        for (int s = 0; s < 16; ++s) { a += s_psum[tid * 16 + s]; b += s_psq[tid * 16 + s]; }
        const int bflat = (blockIdx.z * gridDim.y + blockIdx.y) * gridDim.x + blockIdx.x;
        psum[(size_t)bflat * COUT + tid] = a;
        psq [(size_t)bflat * COUT + tid] = b;
    }
}

// One block per channel; fixed-order strided + tree reduction (deterministic).
__global__ __launch_bounds__(256)
void reduce_stats(const float* __restrict__ psum, const float* __restrict__ psq,
                  float* __restrict__ gsum, float* __restrict__ gsq, int cout)
{
    __shared__ float ssum[256], ssq[256];
    const int c = blockIdx.x, t = threadIdx.x;
    float a = 0.0f, b = 0.0f;
    for (int i = t; i < NBLK; i += 256) {
        a += psum[(size_t)i * cout + c];
        b += psq [(size_t)i * cout + c];
    }
    ssum[t] = a; ssq[t] = b; __syncthreads();
    for (int s = 128; s > 0; s >>= 1) {
        if (t < s) { ssum[t] += ssum[t + s]; ssq[t] += ssq[t + s]; }
        __syncthreads();
    }
    if (t == 0) { gsum[c] = ssum[0]; gsq[c] = ssq[0]; }
}

// BN (pre-reduced stats) + LeakyReLU. 8-wide vectorized.
template<int COUT, bool F32OUT>
__global__ __launch_bounds__(256)
void bn_lrelu(const _Float16* __restrict__ y, const float* __restrict__ gsum,
              const float* __restrict__ gsq, const float* __restrict__ g,
              const float* __restrict__ beta, _Float16* __restrict__ out16,
              float* __restrict__ out32)
{
    const size_t total8 = (size_t)BN_ * COUT * HW_ / 8;
    size_t i = (size_t)blockIdx.x * blockDim.x + threadIdx.x;
    if (i >= total8) return;
    size_t base = i * 8;
    int c = (int)((base / HW_) % COUT);
    const float invM  = 1.0f / (float)(BN_ * HW_);
    float mu    = gsum[c] * invM;
    float var   = gsq[c] * invM - mu * mu;
    float scale = g[c] * rsqrtf(var + EPS_);
    float shift = beta[c] - mu * scale;
    v8h yv = *(const v8h*)(y + base);
    if (F32OUT) {
        #pragma unroll
        for (int j = 0; j < 8; ++j) {
            float v = (float)yv[j] * scale + shift;
            out32[base + j] = (v >= 0.0f) ? v : SLOPE_ * v;
        }
    } else {
        v8h ov;
        #pragma unroll
        for (int j = 0; j < 8; ++j) {
            float v = (float)yv[j] * scale + shift;
            ov[j] = (_Float16)((v >= 0.0f) ? v : SLOPE_ * v);
        }
        *(v8h*)(out16 + base) = ov;
    }
}

// Anchor/positive gathers, 9x9 patch extraction, recon matmul (tiny).
__global__ __launch_bounds__(256)
void gather_recon(const float* __restrict__ x, const int* __restrict__ anchors,
                  const int* __restrict__ positives, const float* __restrict__ z,
                  const float* __restrict__ rw, const float* __restrict__ rb,
                  float* __restrict__ out)
{
    constexpr int ZC = 128;
    constexpr size_t ZSIZE = (size_t)BN_ * ZC * HW_;
    constexpr int PPP = CIN0 * PP_ * PP_;            // 243
    const size_t XA = ZSIZE;
    const size_t XR = XA + (size_t)BN_ * PPP;
    const size_t ZA = XR + (size_t)BN_ * PPP;
    const size_t ZP = ZA + (size_t)BN_ * ZC;

    __shared__ float s_za[BN_ * ZC];
    __shared__ int   s_ahw[BN_ * 2];
    const int tid = threadIdx.x;
    if (tid < BN_ * 2) {
        int nn = tid >> 1, d = tid & 1;
        s_ahw[tid] = anchors[(nn * NS_ + (NS_ - 1)) * 2 + d];
    }
    __syncthreads();

    for (int i = tid; i < BN_ * ZC; i += 256) {
        int nn = i / ZC, c = i % ZC;
        int ah = s_ahw[nn * 2], aw = s_ahw[nn * 2 + 1];
        float v = z[(((size_t)nn * ZC + c) * H_ + ah) * W_ + aw];
        s_za[i] = v;
        out[ZA + i] = v;
        int ph = positives[(nn * NS_ + (NS_ - 1)) * 2 + 0];
        int pw = positives[(nn * NS_ + (NS_ - 1)) * 2 + 1];
        out[ZP + i] = z[(((size_t)nn * ZC + c) * H_ + ph) * W_ + pw];
    }
    __syncthreads();

    for (int j = tid; j < BN_ * PPP; j += 256) {
        int nn = j / PPP, t = j % PPP;
        int c = t / (PP_ * PP_), rr = (t % (PP_ * PP_)) / PP_, cc = t % PP_;
        int ah = s_ahw[nn * 2], aw = s_ahw[nn * 2 + 1];
        out[XA + j] = x[(((size_t)nn * CIN0 + c) * H_ + (ah - HALFP + rr)) * W_ + (aw - HALFP + cc)];
        float acc = rb[t];
        #pragma unroll 4
        for (int ch = 0; ch < ZC; ++ch)
            acc += s_za[nn * ZC + ch] * rw[(size_t)t * ZC + ch];
        out[XR + j] = acc;
    }
}

// ---------------------------------------------------------------------------
extern "C" void kernel_launch(void* const* d_in, const int* in_sizes, int n_in,
                              void* d_out, int out_size, void* d_ws, size_t ws_size,
                              hipStream_t stream)
{
    (void)in_sizes; (void)n_in; (void)out_size; (void)ws_size;
    const float* x   = (const float*)d_in[0];
    const int*   anc = (const int*)d_in[1];
    const int*   pos = (const int*)d_in[2];
    const float* c1w = (const float*)d_in[3];  const float* c1b = (const float*)d_in[4];
    const float* g1  = (const float*)d_in[5];  const float* b1  = (const float*)d_in[6];
    const float* c2w = (const float*)d_in[7];  const float* c2b = (const float*)d_in[8];
    const float* g2  = (const float*)d_in[9];  const float* b2  = (const float*)d_in[10];
    const float* c3w = (const float*)d_in[11]; const float* c3b = (const float*)d_in[12];
    const float* g3  = (const float*)d_in[13]; const float* b3  = (const float*)d_in[14];
    const float* c4w = (const float*)d_in[15]; const float* c4b = (const float*)d_in[16];
    const float* g4  = (const float*)d_in[17]; const float* b4  = (const float*)d_in[18];
    const float* rw  = (const float*)d_in[19]; const float* rb  = (const float*)d_in[20];
    float* out = (float*)d_out;

    char* ws = (char*)d_ws;
    size_t off = 0;
    auto take = [&](size_t bytes) -> char* {
        char* p = ws + off;
        off = (off + bytes + 255) & ~(size_t)255;
        return p;
    };
    _Float16* ybuf = (_Float16*)take((size_t)BN_ * 128 * HW_ * 2);  // conv out (max 128ch)
    _Float16* actA = (_Float16*)take((size_t)BN_ *  32 * HW_ * 2);  // x(f16) / act2
    _Float16* actB = (_Float16*)take((size_t)BN_ *  64 * HW_ * 2);  // act1 / act3
    _Float16* wp1  = (_Float16*)take((size_t)1 *  1 * 512 * 2);
    _Float16* wp2  = (_Float16*)take((size_t)2 *  5 * 512 * 2);
    _Float16* wp3  = (_Float16*)take((size_t)4 *  9 * 512 * 2);
    _Float16* wp4  = (_Float16*)take((size_t)8 * 18 * 512 * 2);
    float* psum = (float*)take((size_t)NBLK * 128 * 4);
    float* psq  = (float*)take((size_t)NBLK * 128 * 4);
    float* gsum = (float*)take(128 * 4);
    float* gsq  = (float*)take(128 * 4);

    // Pre-pack conv weights into WMMA B-fragment order (channel-innermost K).
    pack_weights<<<(1 * 1 * 512 + 255) / 256, 256, 0, stream>>>(c1w, wp1, 16, 3, 1);
    pack_weights<<<(2 * 5 * 512 + 255) / 256, 256, 0, stream>>>(c2w, wp2, 32, 16, 5);
    pack_weights<<<(4 * 9 * 512 + 255) / 256, 256, 0, stream>>>(c3w, wp3, 64, 32, 9);
    pack_weights<<<(8 * 18 * 512 + 255) / 256, 256, 0, stream>>>(c4w, wp4, 128, 64, 18);

    const size_t n0 = (size_t)BN_ * CIN0 * HW_;
    cast_f32_to_f16<<<(unsigned)((n0 + 255) / 256), 256, 0, stream>>>(x, actA, n0);

    const dim3 cgrid(W_ / WTILE, H_, BN_);
    auto bnblocks = [](int cout) -> unsigned {
        return (unsigned)(((size_t)BN_ * cout * HW_ / 8 + 255) / 256);
    };

    // Layer 1: actA(3ch f16) -> ybuf -> actB(16ch f16)
    conv_wmma<3, 16><<<cgrid, 256, 0, stream>>>(actA, wp1, c1b, ybuf, psum, psq);
    reduce_stats<<<16, 256, 0, stream>>>(psum, psq, gsum, gsq, 16);
    bn_lrelu<16, false><<<bnblocks(16), 256, 0, stream>>>(ybuf, gsum, gsq, g1, b1, actB, nullptr);

    // Layer 2: actB(16) -> ybuf -> actA(32)
    conv_wmma<16, 32><<<cgrid, 256, 0, stream>>>(actB, wp2, c2b, ybuf, psum, psq);
    reduce_stats<<<32, 256, 0, stream>>>(psum, psq, gsum, gsq, 32);
    bn_lrelu<32, false><<<bnblocks(32), 256, 0, stream>>>(ybuf, gsum, gsq, g2, b2, actA, nullptr);

    // Layer 3: actA(32) -> ybuf -> actB(64)
    conv_wmma<32, 64><<<cgrid, 256, 0, stream>>>(actA, wp3, c3b, ybuf, psum, psq);
    reduce_stats<<<64, 256, 0, stream>>>(psum, psq, gsum, gsq, 64);
    bn_lrelu<64, false><<<bnblocks(64), 256, 0, stream>>>(ybuf, gsum, gsq, g3, b3, actB, nullptr);

    // Layer 4: actB(64) -> ybuf -> z (fp32, d_out offset 0)
    conv_wmma<64, 128><<<cgrid, 256, 0, stream>>>(actB, wp4, c4b, ybuf, psum, psq);
    reduce_stats<<<128, 256, 0, stream>>>(psum, psq, gsum, gsq, 128);
    bn_lrelu<128, true><<<bnblocks(128), 256, 0, stream>>>(ybuf, gsum, gsq, g4, b4, nullptr, out);

    // Gathers + tiny recon GEMM.
    gather_recon<<<1, 256, 0, stream>>>(x, anc, pos, out, rw, rb, out);
}
